// molecular_network_75196287418641
// MI455X (gfx1250) — matrix operand order repaired and checked by reading, not compiled, not run on hardware
//
#include <hip/hip_runtime.h>

// ---------------------------------------------------------------------------
// NequIP-style equivariant GNN forward for MI455X (gfx1250, wave32, WMMA).
//
// Fused per-layer edge kernel: radial MLP (WMMA f16 16x16x32, f32 accum) with
// the tensor-product scatter consumed tile-by-tile in LDS, so the huge
// per-edge weight tensor tpw (up to 767 MB/layer) never touches HBM.
// ---------------------------------------------------------------------------

typedef _Float16 half_t;
typedef __attribute__((ext_vector_type(16))) _Float16 v16h;
typedef __attribute__((ext_vector_type(8)))  float    v8f;

#define N_NODES 2048
#define N_EDGES 12288
#define N_GRAPHS 64

__device__ __forceinline__ float fast_sigmoid(float v) {
  // v_rcp_f32 instead of full IEEE division ladder
  return __builtin_amdgcn_rcpf(1.f + __expf(-v));
}
__device__ __forceinline__ float fast_silu(float v) {
  return v * fast_sigmoid(v);
}

// ------------------------- tensor-product metadata -------------------------

struct TPPath {
  short l1, l2, lo, m1, mo;
  int o1, oo, woff, w3off;
  float alpha;
};
struct TPMeta { int npaths; TPPath p[30]; };

#define A40  0.15811388f   // 1/sqrt(40)
#define A60  0.12909944f   // 1/sqrt(60)
#define A80  0.11180340f   // 1/sqrt(80)
#define A100 0.10000000f   // 1/sqrt(100)
#define A120 0.09128709f   // 1/sqrt(120)
#define A8   0.35355339f   // 1/sqrt(8)

__constant__ TPMeta TPM[4] = {
  // ---- layer 0: in [(8,0,+)], out [(20,0,+),(40,0,+),(20,1,-),(20,2,+)] ----
  {4, {
    {0,0,0, 8,20, 0,  0,    0,   0, A8},
    {0,0,0, 8,40, 0,  20,   160, 0, A8},
    {0,1,1, 8,20, 0,  60,   480, 1, A8},
    {0,2,2, 8,20, 0,  120,  640, 10, A8},
  }},
  // ---- layer 1: in [(20,0,+),(20,1,-),(20,2,+)] ----
  {18, {
    {0,0,0,20,20,  0,  0,    0,    0,   A60},
    {0,0,0,20,80,  0,  20,   400,  0,   A60},
    {0,1,1,20,20,  0,  100,  2000, 1,   A80},
    {0,2,2,20,20,  0,  320,  2400, 10,  A80},
    {1,0,1,20,20, 20,  100,  2800, 35,  A80},
    {1,1,0,20,20, 20,  0,    3200, 44,  A60},
    {1,1,0,20,80, 20,  20,   3600, 44,  A60},
    {1,1,1,20,20, 20,  160,  5200, 53,  A40},
    {1,1,2,20,20, 20,  320,  5600, 80,  A80},
    {1,2,1,20,20, 20,  100,  6000, 125, A80},
    {1,2,2,20,20, 20,  220,  6400, 170, A40},
    {2,0,2,20,20, 80,  320,  6800, 245, A80},
    {2,1,1,20,20, 80,  100,  7200, 270, A80},
    {2,1,2,20,20, 80,  220,  7600, 315, A40},
    {2,2,0,20,20, 80,  0,    8000, 390, A60},
    {2,2,0,20,80, 80,  20,   8400, 390, A60},
    {2,2,1,20,20, 80,  160,  10000,415, A40},
    {2,2,2,20,20, 80,  320,  10400,490, A80},
  }},
  // ---- layer 2: in [(20,0,+),(20,1,-),(20,1,+),(20,2,-),(20,2,+)] ----
  {30, {
    {0,0,0,20,20,   0, 20,   0,    0,   A60},
    {0,0,0,20,80,   0, 40,   400,  0,   A60},
    {0,1,1,20,20,   0, 120,  2000, 1,   A120},
    {0,2,2,20,20,   0, 340,  2400, 10,  A120},
    {1,0,1,20,20,  20, 120,  2800, 35,  A120},
    {1,1,0,20,20,  20, 20,   3200, 44,  A60},
    {1,1,0,20,80,  20, 40,   3600, 44,  A60},
    {1,1,1,20,20,  20, 180,  5200, 53,  A100},
    {1,1,2,20,20,  20, 340,  5600, 80,  A120},
    {1,2,1,20,20,  20, 120,  6000, 125, A120},
    {1,2,2,20,20,  20, 240,  6400, 170, A100},
    {1,0,1,20,20,  80, 180,  6800, 35,  A100},
    {1,1,0,20,20,  80, 0,    7200, 44,  A40},
    {1,1,1,20,20,  80, 120,  7600, 53,  A120},
    {1,1,2,20,20,  80, 240,  8000, 80,  A100},
    {1,2,1,20,20,  80, 180,  8400, 125, A100},
    {1,2,2,20,20,  80, 340,  8800, 170, A120},
    {2,0,2,20,20, 140, 240,  9200, 245, A100},
    {2,1,1,20,20, 140, 180,  9600, 270, A100},
    {2,1,2,20,20, 140, 340,  10000,315, A120},
    {2,2,0,20,20, 140, 0,    10400,390, A40},
    {2,2,1,20,20, 140, 120,  10800,415, A120},
    {2,2,2,20,20, 140, 240,  11200,490, A100},
    {2,0,2,20,20, 240, 340,  11600,245, A120},
    {2,1,1,20,20, 240, 120,  12000,270, A120},
    {2,1,2,20,20, 240, 240,  12400,315, A100},
    {2,2,0,20,20, 240, 20,   12800,390, A60},
    {2,2,0,20,80, 240, 40,   13200,390, A60},
    {2,2,1,20,20, 240, 180,  14800,415, A100},
    {2,2,2,20,20, 240, 340,  15200,490, A120},
  }},
  // ---- layer 3: in 6 blocks, out [(4,0,+)] ----
  {3, {
    {0,0,0,20,4,  20, 0,   0,   0,  A60},
    {1,1,0,20,4,  40, 0,   80,  44, A60},
    {2,2,0,20,4, 260, 0,   160, 390,A60},
  }},
};

struct GateMeta { int nscal; int scal_m[2]; int scal_tanh[2]; int ngates; int ngated; int gm[4]; int gd[4]; };
__constant__ GateMeta GM[3] = {
  {1, {20,0},  {0,0}, 40, 2, {20,20, 0, 0}, {3,5,0,0}},
  {1, {20,0},  {0,0}, 80, 4, {20,20,20,20}, {3,3,5,5}},
  {2, {20,20}, {1,0}, 80, 4, {20,20,20,20}, {3,3,5,5}},
};

// ------------------------- Wigner-3j init (exact port) ----------------------

struct C2 { double re, im; };
__device__ inline C2 cmul(C2 a, C2 b){ return {a.re*b.re - a.im*b.im, a.re*b.im + a.im*b.re}; }
__device__ inline double dfact(int n){ double r = 1.0; for (int i = 2; i <= n; ++i) r *= (double)i; return r; }

__device__ double cgcoef(int j1,int m1,int j2,int m2,int j3,int m3){
  if (m1 + m2 != m3) return 0.0;
  double pref = sqrt((double)(2*j3+1) * dfact(j1+j2-j3)*dfact(j1-j2+j3)*dfact(-j1+j2+j3)/dfact(j1+j2+j3+1)
                     * dfact(j3+m3)*dfact(j3-m3)*dfact(j1-m1)*dfact(j1+m1)*dfact(j2-m2)*dfact(j2+m2));
  double s = 0.0;
  for (int k = 0; k <= j1+j2-j3; ++k) {
    int t0=k, t1=j1+j2-j3-k, t2=j1-m1-k, t3=j2+m2-k, t4=j3-j2+m1+k, t5=j3-j1-m2+k;
    if (t0<0||t1<0||t2<0||t3<0||t4<0||t5<0) continue;
    double d = dfact(t0)*dfact(t1)*dfact(t2)*dfact(t3)*dfact(t4)*dfact(t5);
    s += ((k & 1) ? -1.0 : 1.0) / d;
  }
  return pref * s;
}

__device__ void make_q(int l, C2 q[5][5]){
  for (int i = 0; i < 5; ++i) for (int j = 0; j < 5; ++j) q[i][j] = {0.0, 0.0};
  const double is2 = 0.7071067811865476;
  for (int mm = -l; mm < 0; ++mm) { q[l+mm][l-mm] = {is2, 0.0}; q[l+mm][l+mm] = {0.0, -is2}; }
  q[l][l] = {1.0, 0.0};
  for (int mm = 1; mm <= l; ++mm) {
    double sg = (mm & 1) ? -1.0 : 1.0;
    q[l+mm][l+mm] = {sg*is2, 0.0};
    q[l+mm][l-mm] = {0.0, sg*is2};
  }
  C2 f = (l == 0) ? C2{1.0, 0.0} : (l == 1 ? C2{0.0, -1.0} : C2{-1.0, 0.0});  // (-i)^l
  for (int i = 0; i < 5; ++i) for (int j = 0; j < 5; ++j) q[i][j] = cmul(f, q[i][j]);
}

__global__ void w3j_init(float* __restrict__ w3j){
  const int t = threadIdx.x;
  const int L1t[15] = {0,0,0,1,1,1,1,1,1,2,2,2,2,2,2};
  const int L2t[15] = {0,1,2,0,1,1,1,2,2,0,1,1,2,2,2};
  const int L3t[15] = {0,1,2,1,0,1,2,1,2,2,1,2,0,1,2};
  const int OFFt[15]= {0,1,10,35,44,53,80,125,170,245,270,315,390,415,490};
  if (t >= 15) return;
  const int l1 = L1t[t], l2 = L2t[t], l3 = L3t[t];
  const int d1 = 2*l1+1, d2 = 2*l2+1, d3 = 2*l3+1;
  C2 q1[5][5], q2[5][5], q3[5][5];
  make_q(l1, q1); make_q(l2, q2); make_q(l3, q3);
  double Wr[5][5][5], Wi[5][5][5];
  double nr = 0.0, ni = 0.0;
  for (int i = 0; i < d1; ++i) for (int j = 0; j < d2; ++j) for (int k = 0; k < d3; ++k) {
    C2 acc = {0.0, 0.0};
    for (int a = 0; a < d1; ++a) {
      int m1 = a - l1;
      for (int b = 0; b < d2; ++b) {
        int m2 = b - l2, m3 = m1 + m2;
        if (m3 < -l3 || m3 > l3) continue;
        double cgv = cgcoef(l1, m1, l2, m2, l3, m3);
        if (cgv == 0.0) continue;
        C2 c3 = {q3[l3+m3][k].re, -q3[l3+m3][k].im};   // conj
        C2 tt = cmul(q1[a][i], cmul(q2[b][j], c3));
        acc.re += cgv * tt.re; acc.im += cgv * tt.im;
      }
    }
    Wr[i][j][k] = acc.re; Wi[i][j][k] = acc.im;
    nr += acc.re * acc.re; ni += acc.im * acc.im;
  }
  const bool useR = (nr >= ni);
  const double inv = 1.0 / sqrt(useR ? nr : ni);
  for (int i = 0; i < d1; ++i) for (int j = 0; j < d2; ++j) for (int k = 0; k < d3; ++k)
    w3j[OFFt[t] + (i*d2 + j)*d3 + k] = (float)((useR ? Wr[i][j][k] : Wi[i][j][k]) * inv);
}

// ------------------------- edge geometry -----------------------------------

__global__ void edge_geom(const float* __restrict__ pos, const int* __restrict__ esrc,
                          const int* __restrict__ edst, float* __restrict__ sh,
                          float* __restrict__ demb){
  int e = blockIdx.x * blockDim.x + threadIdx.x;
  if (e >= N_EDGES) return;
  int s = esrc[e], d = edst[e];
  float vx = pos[d*3+0]-pos[s*3+0], vy = pos[d*3+1]-pos[s*3+1], vz = pos[d*3+2]-pos[s*3+2];
  float r = sqrtf(vx*vx + vy*vy + vz*vz);
  float inv = __builtin_amdgcn_rcpf(r);
  float x = vx*inv, y = vy*inv, z = vz*inv;
  const float SQ3 = 1.7320508f, SQ15 = 3.8729833f, SQ5H = 1.1180340f, SQ15H = 1.9364916f;
  float* o = sh + (size_t)e * 9;
  o[0] = 1.f;
  o[1] = SQ3*y; o[2] = SQ3*z; o[3] = SQ3*x;
  o[4] = SQ15*x*y; o[5] = SQ15*y*z; o[6] = SQ5H*(2.f*z*z - x*x - y*y);
  o[7] = SQ15*x*z; o[8] = SQ15H*(x*x - y*y);
  float* de = demb + (size_t)e * 20;
  const float step = 5.f / 21.f;
  const float istep = 21.f / 5.f;
  for (int i = 0; i < 20; ++i) {
    float diff = (r - step*(float)(i+1)) * istep;
    float x2 = diff * diff;
    de[i] = (x2 < 1.f) ? 8.433573f * __expf(-__builtin_amdgcn_rcpf(1.f - x2)) : 0.f;  // 1.14136*e^2
  }
}

// ------------------------- weight packing (f16, B-layout) -------------------
// B packing uses the same k order as the A fragments: lane L, element j -> N = L&15, K = j + 16*(L>=16)

__global__ void pack_w1(const float* __restrict__ w1, half_t* __restrict__ w1p){
  int t = blockIdx.x * blockDim.x + threadIdx.x;
  if (t >= 7*32) return;
  int nt = t >> 5, lane = t & 31;
  for (int j = 0; j < 16; ++j) {
    int K = j + 16*(lane >= 16);
    int N = nt*16 + (lane & 15);
    float v = (K < 20 && N < 100) ? w1[K*100 + N] * 0.22360680f : 0.f;  // 1/sqrt(20)
    w1p[(nt*32 + lane)*16 + j] = (half_t)v;
  }
}

__global__ void pack_w2(const float* __restrict__ w2, half_t* __restrict__ w2p, int numel){
  int t = blockIdx.x * blockDim.x + threadIdx.x;
  int NT = numel >> 4;
  if (t >= NT * 128) return;
  int nt = t >> 7, kt = (t >> 5) & 3, lane = t & 31;
  for (int j = 0; j < 16; ++j) {
    int K = kt*32 + j + 16*(lane >= 16);
    int N = nt*16 + (lane & 15);
    float v = (K < 100) ? w2[(size_t)K*numel + N] * 0.1f : 0.f;        // 1/sqrt(100)
    w2p[(((size_t)nt*4 + kt)*32 + lane)*16 + j] = (half_t)v;
  }
}

// ------------------------- fused conv kernel --------------------------------
// 1 wave / block, 16 edges per wave. A layout: lane L elem j -> M=L&15,
// K = j + 16*(L>=16). C layout: lane L, VGPR r -> N=L&15, M=r+8*(L>=16).

template<int IN, int OUT, int TMAX>
__global__ __launch_bounds__(32)
void conv_kernel(const float* __restrict__ xprev, const float* __restrict__ sh,
                 const float* __restrict__ demb, const half_t* __restrict__ w1p,
                 const half_t* __restrict__ w2p, const float* __restrict__ w3j,
                 const int* __restrict__ esrc, const int* __restrict__ edst,
                 float* __restrict__ xconv, int layer)
{
  __shared__ float  xrow[16][IN];
  __shared__ float  shs[16][9];
  __shared__ float  Ts[16][TMAX];
  __shared__ float  efs[16][OUT];
  __shared__ half_t hs[16][128];
  __shared__ float  cgs[128];
  __shared__ int    dsts[16], srcs[16];

  const int lane = threadIdx.x;
  const int m  = lane & 15;
  const int hi = lane >> 4;
  const int e0 = blockIdx.x * 16;

  if (lane < 16) { dsts[lane] = edst[e0 + lane]; srcs[lane] = esrc[e0 + lane]; }
  for (int i = lane; i < 16*OUT; i += 32) (&efs[0][0])[i] = 0.f;
  __syncthreads();

  for (int i = lane; i < 16*IN; i += 32) { int e = i / IN, d = i - e*IN; xrow[e][d] = xprev[(size_t)dsts[e]*IN + d]; }
  for (int i = lane; i < 16*9;  i += 32) { int e = i / 9,  d = i - e*9;  shs[e][d]  = sh[(size_t)(e0 + e)*9 + d]; }

  // A fragment: distance embedding (K padded 20->32 with zeros)
  v16h a;
  #pragma unroll
  for (int j = 0; j < 16; ++j) {
    int K = j + 16*hi;
    a[j] = (K < 20) ? (half_t)demb[(size_t)(e0 + m)*20 + K] : (half_t)0.f;
  }

  // h = silu(demb @ W1) : 16 edges x 100 (stored padded to 128, f16)
  for (int nt = 0; nt < 7; ++nt) {
    v16h b = *(const v16h*)(w1p + (nt*32 + lane)*16);
    v8f c = {0.f,0.f,0.f,0.f,0.f,0.f,0.f,0.f};
    c = __builtin_amdgcn_wmma_f32_16x16x32_f16(false, a, false, b, (short)0, c, false, false);
    #pragma unroll
    for (int r = 0; r < 8; ++r) {
      hs[r + 8*hi][nt*16 + m] = (half_t)fast_silu(c[r]);
    }
  }
  for (int i = lane; i < 256; i += 32) hs[i >> 4][112 + (i & 15)] = (half_t)0.f;
  __syncthreads();

  // A fragments for h (K = 0..127, 4 k-tiles)
  v16h ha[4];
  #pragma unroll
  for (int kt = 0; kt < 4; ++kt)
    #pragma unroll
    for (int j = 0; j < 16; ++j)
      ha[kt][j] = hs[m][kt*32 + j + 16*hi];

  const TPMeta& M = TPM[layer];
  for (int p = 0; p < M.npaths; ++p) {
    const TPPath P = M.p[p];
    const int d1 = 2*P.l1 + 1, d2 = 2*P.l2 + 1, dlo = 2*P.lo + 1;
    __syncthreads();
    for (int i = lane; i < d1*d2*dlo; i += 32) cgs[i] = w3j[P.w3off + i];
    __syncthreads();
    {   // T[e][u*dlo+k] = sum_{i,j} W3J[i,j,k] * x_dst[e, o1+u*d1+i] * sh[e, l2^2+j]
      const int uh = P.m1 >> 1;
      const int sho = P.l2 * P.l2;
      for (int u = hi*uh; u < hi*uh + uh; ++u)
        for (int k = 0; k < dlo; ++k) {
          float acc = 0.f;
          for (int i = 0; i < d1; ++i) {
            float av = xrow[m][P.o1 + u*d1 + i];
            for (int j = 0; j < d2; ++j)
              acc += cgs[(i*d2 + j)*dlo + k] * av * shs[m][sho + j];
          }
          Ts[m][u*dlo + k] = acc;
        }
    }
    __syncthreads();
    const int ntiles = (P.m1 * P.mo) >> 4;       // path w-range is 16-aligned
    const int nt0 = P.woff >> 4;
    for (int nt = 0; nt < ntiles; ++nt) {
      v8f c = {0.f,0.f,0.f,0.f,0.f,0.f,0.f,0.f};
      #pragma unroll
      for (int kt = 0; kt < 4; ++kt) {
        v16h b = *(const v16h*)(w2p + ((size_t)((nt0 + nt)*4 + kt)*32 + lane)*16);
        c = __builtin_amdgcn_wmma_f32_16x16x32_f16(false, ha[kt], false, b, (short)0, c, false, false);
      }
      const int loc = nt*16 + m;                 // column within path
      const int u = loc / P.mo, o = loc - u*P.mo;
      const int base = P.oo + o*dlo;
      #pragma unroll
      for (int r = 0; r < 8; ++r) {
        const int e = r + 8*hi;
        const float wv = c[r] * P.alpha;
        for (int k = 0; k < dlo; ++k)
          atomicAdd(&efs[e][base + k], wv * Ts[e][u*dlo + k]);  // ds_add_f32
      }
    }
  }
  __syncthreads();

  const float invs = 0.40824829f;  // 1/sqrt(NUM_NEIGHBORS=6)
  for (int i = lane; i < 16*OUT; i += 32) {
    int e = i / OUT, d = i - e*OUT;
    atomicAdd(&xconv[(size_t)srcs[e]*OUT + d], efs[e][d] * invs);
  }
}

// ------------------------- gate + final pooling -----------------------------

__global__ void gate_kernel(const float* __restrict__ xin, float* __restrict__ xout,
                            int layer, int in_dim, int out_dim)
{
  int n = blockIdx.x * blockDim.x + threadIdx.x;
  if (n >= N_NODES) return;
  const GateMeta& G = GM[layer];
  const float* xi = xin + (size_t)n * in_dim;
  float* xo = xout + (size_t)n * out_dim;
  int o = 0, oo = 0;
  for (int s = 0; s < G.nscal; ++s) {
    for (int t = 0; t < G.scal_m[s]; ++t) {
      float v = xi[o + t];
      xo[oo + t] = G.scal_tanh[s] ? tanhf(v) : fast_silu(v);
    }
    o += G.scal_m[s]; oo += G.scal_m[s];
  }
  int go = o;
  o += G.ngates;
  int gidx = 0;
  for (int g = 0; g < G.ngated; ++g) {
    int mm = G.gm[g], dd = G.gd[g];
    for (int u = 0; u < mm; ++u) {
      float gv = fast_sigmoid(xi[go + gidx + u]);
      for (int k = 0; k < dd; ++k)
        xo[oo + u*dd + k] = xi[o + u*dd + k] * gv;
    }
    o += mm*dd; oo += mm*dd; gidx += mm;
  }
}

__global__ void final_reduce(const float* __restrict__ x, const int* __restrict__ batch,
                             float* __restrict__ out)
{
  int n = blockIdx.x * blockDim.x + threadIdx.x;
  if (n >= N_NODES) return;
  int b = batch[n];
  for (int d = 0; d < 4; ++d) atomicAdd(&out[b*4 + d], x[n*4 + d]);
}

// ------------------------- launcher -----------------------------------------

extern "C" void kernel_launch(void* const* d_in, const int* in_sizes, int n_in,
                              void* d_out, int out_size, void* d_ws, size_t ws_size,
                              hipStream_t stream)
{
  (void)in_sizes; (void)n_in; (void)out_size; (void)ws_size;
  const float* node_x = (const float*)d_in[0];
  const float* pos    = (const float*)d_in[1];
  const int*   esrc   = (const int*)d_in[2];
  const int*   edst   = (const int*)d_in[3];
  const int*   batch  = (const int*)d_in[4];
  const float* w1[4]  = {(const float*)d_in[5], (const float*)d_in[7], (const float*)d_in[9],  (const float*)d_in[11]};
  const float* w2[4]  = {(const float*)d_in[6], (const float*)d_in[8], (const float*)d_in[10], (const float*)d_in[12]};

  // workspace layout (floats first; f16 region 256B-aligned after)
  float* wsf   = (float*)d_ws;
  float* w3j   = wsf;                                   // 615 used (1024 reserved)
  float* sh    = wsf + 1024;                            // E*9
  float* demb  = sh + (size_t)N_EDGES * 9;              // E*20
  float* xconv = demb + (size_t)N_EDGES * 20;           // NN*448 (max conv out 440)
  float* xg    = xconv + (size_t)N_NODES * 448;         // NN*368 (max gated 360)
  float* fend  = xg + (size_t)N_NODES * 368;            // total floats = multiple of 64
  half_t* w1p  = (half_t*)fend;                         // 4 * 7*512 halves
  half_t* w2p  = w1p + 4*3584;                          // 1715 tiles * 2048 halves

  const int numel[4]    = {800, 10800, 15600, 240};
  const size_t w2off[4] = {0, (size_t)50*2048, (size_t)725*2048, (size_t)1700*2048};

  w3j_init<<<1, 32, 0, stream>>>(w3j);
  edge_geom<<<(N_EDGES + 255)/256, 256, 0, stream>>>(pos, esrc, edst, sh, demb);
  for (int l = 0; l < 4; ++l) {
    pack_w1<<<1, 256, 0, stream>>>(w1[l], w1p + l*3584);
    int nth = (numel[l] >> 4) * 128;
    pack_w2<<<(nth + 255)/256, 256, 0, stream>>>(w2[l], w2p + w2off[l], numel[l]);
  }

  // layer 0
  hipMemsetAsync(xconv, 0, (size_t)N_NODES*220*sizeof(float), stream);
  conv_kernel<8,220,40><<<N_EDGES/16, 32, 0, stream>>>(node_x, sh, demb, w1p, w2p + w2off[0], w3j, esrc, edst, xconv, 0);
  gate_kernel<<<(N_NODES + 255)/256, 256, 0, stream>>>(xconv, xg, 0, 220, 180);
  // layer 1
  hipMemsetAsync(xconv, 0, (size_t)N_NODES*420*sizeof(float), stream);
  conv_kernel<180,420,100><<<N_EDGES/16, 32, 0, stream>>>(xg, sh, demb, w1p + 3584, w2p + w2off[1], w3j, esrc, edst, xconv, 1);
  gate_kernel<<<(N_NODES + 255)/256, 256, 0, stream>>>(xconv, xg, 1, 420, 340);
  // layer 2
  hipMemsetAsync(xconv, 0, (size_t)N_NODES*440*sizeof(float), stream);
  conv_kernel<340,440,100><<<N_EDGES/16, 32, 0, stream>>>(xg, sh, demb, w1p + 2*3584, w2p + w2off[2], w3j, esrc, edst, xconv, 2);
  gate_kernel<<<(N_NODES + 255)/256, 256, 0, stream>>>(xconv, xg, 2, 440, 360);
  // layer 3 (output tensor product)
  hipMemsetAsync(xconv, 0, (size_t)N_NODES*4*sizeof(float), stream);
  conv_kernel<360,4,20><<<N_EDGES/16, 32, 0, stream>>>(xg, sh, demb, w1p + 3*3584, w2p + w2off[3], w3j, esrc, edst, xconv, 3);

  // per-graph pooling
  hipMemsetAsync(d_out, 0, (size_t)N_GRAPHS*4*sizeof(float), stream);
  final_reduce<<<(N_NODES + 255)/256, 256, 0, stream>>>(xconv, batch, (float*)d_out);
}
